// AttentionMultiLayersDecoder_14181982012225
// MI455X (gfx1250) — matrix operand order repaired and checked by reading, not compiled, not run on hardware
//
#include <hip/hip_runtime.h>
#include <hip/hip_bf16.h>

#define BB 32
#define TT 64
#define SS 64
#define EE 256
#define HH 512
#define AAT 512
#define DVV 32000
#define K0 (HH + EE + HH)   // 1280 : [context | x_t | h0]

typedef float  v2f   __attribute__((ext_vector_type(2)));
typedef float  v8f   __attribute__((ext_vector_type(8)));
typedef __bf16 v16bf __attribute__((ext_vector_type(16)));

__device__ __forceinline__ float sigmoidf_(float x) { return 1.f / (1.f + __expf(-x)); }

// ---------------------------------------------------------------------------
// Generic f32 WMMA GEMM: C[M,N] = A[M,K] @ B[K,N] + bias[N]
// One wave per 16x16 tile (used for the big M=2048 enc_proj GEMM).
// ---------------------------------------------------------------------------
__global__ void gemm_f32_wmma(const float* __restrict__ A, const float* __restrict__ Bm,
                              const float* __restrict__ bias, float* __restrict__ C,
                              int M, int N, int K) {
  const int lane = threadIdx.x & 31;
  const int wave = threadIdx.x >> 5;
  const int tile = blockIdx.x * (blockDim.x >> 5) + wave;
  const int ntn  = N >> 4;
  const int tm   = tile / ntn;
  const int tn   = tile - tm * ntn;
  if (tm * 16 >= M) return;
  const int l16 = lane & 15;
  const int h   = lane >> 4;
  const int m   = tm * 16 + l16;
  const int n   = tn * 16 + l16;
  const float* arow = A + (size_t)m * K;
  v8f acc = {};
  for (int k0 = 0; k0 < K; k0 += 4) {
    v2f a, b;
    a.x = arow[k0 + 2 * h];
    a.y = arow[k0 + 2 * h + 1];
    b.x = Bm[(size_t)(k0 + 2 * h) * N + n];
    b.y = Bm[(size_t)(k0 + 2 * h + 1) * N + n];
    acc = __builtin_amdgcn_wmma_f32_16x16x4_f32(false, a, false, b, (short)0, acc,
                                                false, false);
  }
  const float bv = bias ? bias[n] : 0.f;
#pragma unroll
  for (int r = 0; r < 8; ++r)
    C[(size_t)(tm * 16 + r + 8 * h) * N + n] = acc[r] + bv;
}

// ---------------------------------------------------------------------------
// M=32 specialized f32 WMMA GEMM: one wave per 16-column strip computes the
// full 32 rows with two accumulators, so every B fragment feeds two WMMAs
// (halves B-side L2 traffic for the q / z0 / z1 GEMMs).
// ---------------------------------------------------------------------------
__global__ void gemm_f32_wmma_m32(const float* __restrict__ A, const float* __restrict__ Bm,
                                  const float* __restrict__ bias, float* __restrict__ C,
                                  int N, int K) {
  const int lane = threadIdx.x & 31;
  const int wave = threadIdx.x >> 5;
  const int tn   = blockIdx.x * (blockDim.x >> 5) + wave;
  if (tn * 16 >= N) return;
  const int l16 = lane & 15;
  const int h   = lane >> 4;
  const int n   = tn * 16 + l16;
  const float* a0row = A + (size_t)l16 * K;         // rows 0..15
  const float* a1row = A + (size_t)(16 + l16) * K;  // rows 16..31
  v8f acc0 = {}, acc1 = {};
  for (int k0 = 0; k0 < K; k0 += 4) {
    v2f a0, a1, b;
    a0.x = a0row[k0 + 2 * h];
    a0.y = a0row[k0 + 2 * h + 1];
    a1.x = a1row[k0 + 2 * h];
    a1.y = a1row[k0 + 2 * h + 1];
    b.x  = Bm[(size_t)(k0 + 2 * h) * N + n];
    b.y  = Bm[(size_t)(k0 + 2 * h + 1) * N + n];
    acc0 = __builtin_amdgcn_wmma_f32_16x16x4_f32(false, a0, false, b, (short)0, acc0,
                                                 false, false);
    acc1 = __builtin_amdgcn_wmma_f32_16x16x4_f32(false, a1, false, b, (short)0, acc1,
                                                 false, false);
  }
  const float bv = bias ? bias[n] : 0.f;
#pragma unroll
  for (int r = 0; r < 8; ++r) {
    C[(size_t)(r + 8 * h) * N + n]      = acc0[r] + bv;
    C[(size_t)(16 + r + 8 * h) * N + n] = acc1[r] + bv;
  }
}

// ---------------------------------------------------------------------------
// Logits GEMM bf16 x bf16 -> f32: out[b,t,:] = h1 @ Wd + bd (M=32,N=32000,K=512)
// One wave per 16-column strip, both 16-row output tiles per wave.
// B tiles (32k x 16n bf16 = 1KB) are staged into a per-wave double-buffered
// LDS tile with async global->LDS b128 copies (ASYNCcnt-tracked, one k-block
// of prefetch ahead of the WMMA pipe).  A is pre-swizzled; NT stores for out.
// ---------------------------------------------------------------------------
__global__ void gemm_logits_bf16(const __bf16* __restrict__ Abuf,
                                 const __bf16* __restrict__ Wdbf,
                                 const float* __restrict__ bd,
                                 float* __restrict__ out, int t) {
  __shared__ __align__(16) __bf16 btile[8][2][32][16];  // [wave][buf][k][n] = 16KB
  const int lane = threadIdx.x & 31;
  const int wave = threadIdx.x >> 5;
  const int tn   = blockIdx.x * (blockDim.x >> 5) + wave;  // 0..1999
  const int l16  = lane & 15;
  const int h    = lane >> 4;
  const int n    = tn * 16 + l16;

  // lane l copies Wd row (kblock*32 + l), columns [tn*16, tn*16+16), 32 bytes
  const uint64_t gbase =
      (uint64_t)(uintptr_t)(Wdbf + (size_t)lane * DVV + (size_t)tn * 16);
  const uint64_t gstride = (uint64_t)32 * DVV * sizeof(__bf16);
  const uint32_t lds0 = (uint32_t)(uintptr_t)&btile[wave][0][lane][0];
  const uint32_t lds1 = (uint32_t)(uintptr_t)&btile[wave][1][lane][0];

  // prime k-block 0 into buffer 0
  asm volatile("global_load_async_to_lds_b128 %0, %1, off\n\t"
               "global_load_async_to_lds_b128 %0, %1, off offset:16"
               :: "v"(lds0), "v"(gbase) : "memory");

  v8f acc0 = {}, acc1 = {};
  for (int kk = 0; kk < HH / 32; ++kk) {
    const int cur = kk & 1;
    if (kk < HH / 32 - 1) {
      const uint64_t gn      = gbase + (uint64_t)(kk + 1) * gstride;
      const uint32_t ldsnext = cur ? lds0 : lds1;
      asm volatile("s_wait_dscnt 0x0\n\t"  // prior reads of next buffer retired
                   "global_load_async_to_lds_b128 %0, %1, off\n\t"
                   "global_load_async_to_lds_b128 %0, %1, off offset:16"
                   :: "v"(ldsnext), "v"(gn) : "memory");
      asm volatile("s_wait_asynccnt 0x2" ::: "memory");  // current buffer ready
    } else {
      asm volatile("s_wait_asynccnt 0x0" ::: "memory");
    }
    v16bf a0 = *(const v16bf*)(Abuf + (((kk * 2 + 0) * 32 + lane) << 4));
    v16bf a1 = *(const v16bf*)(Abuf + (((kk * 2 + 1) * 32 + lane) << 4));
    v16bf b;
#pragma unroll
    for (int j = 0; j < 16; ++j) b[j] = btile[wave][cur][8 * h + j + (j >= 8 ? 8 : 0)][l16];
    acc0 = __builtin_amdgcn_wmma_f32_16x16x32_bf16(false, a0, false, b, (short)0, acc0,
                                                   false, false);
    acc1 = __builtin_amdgcn_wmma_f32_16x16x32_bf16(false, a1, false, b, (short)0, acc1,
                                                   false, false);
  }
  const float bv = bd[n];
#pragma unroll
  for (int r = 0; r < 8; ++r) {
    const int m0 = r + 8 * h;       // batch rows 0..15
    const int m1 = 16 + r + 8 * h;  // batch rows 16..31
    __builtin_nontemporal_store(acc0[r] + bv, &out[((size_t)m0 * TT + t) * DVV + n]);
    __builtin_nontemporal_store(acc1[r] + bv, &out[((size_t)m1 * TT + t) * DVV + n]);
  }
}

// ---------------------------------------------------------------------------
// Attention for one step; builds x0cat = [context | x_t | h0] (K0=1280 cols).
// One workgroup (8 waves) per batch row.
// ---------------------------------------------------------------------------
__global__ void attn_ctx(const float* __restrict__ q, const float* __restrict__ enc_proj,
                         const float* __restrict__ enc, const float* __restrict__ Vv,
                         const float* __restrict__ bV, const float* __restrict__ x_in,
                         const float* __restrict__ h0, int t, float* __restrict__ x0cat) {
  const int b    = blockIdx.x;
  const int tid  = threadIdx.x;  // 256
  const int lane = tid & 31, wave = tid >> 5;
  __shared__ float sc[SS];
  const float* qb = q + (size_t)b * AAT;
  for (int s = wave; s < SS; s += 8) {
    const float* ep  = enc_proj + ((size_t)b * SS + s) * AAT;
    float        acc = 0.f;
    for (int k = lane; k < AAT; k += 32) acc += Vv[k] * tanhf(ep[k] + qb[k]);
#pragma unroll
    for (int off = 16; off; off >>= 1) acc += __shfl_xor(acc, off, 32);
    if (lane == 0) sc[s] = acc + bV[0];
  }
  __syncthreads();
  if (wave == 0) {
    float v0 = sc[lane], v1 = sc[lane + 32];
    float mx = fmaxf(v0, v1);
#pragma unroll
    for (int off = 16; off; off >>= 1) mx = fmaxf(mx, __shfl_xor(mx, off, 32));
    float e0 = __expf(v0 - mx), e1 = __expf(v1 - mx);
    float sm = e0 + e1;
#pragma unroll
    for (int off = 16; off; off >>= 1) sm += __shfl_xor(sm, off, 32);
    const float inv = 1.f / sm;
    sc[lane]      = e0 * inv;
    sc[lane + 32] = e1 * inv;
  }
  __syncthreads();
  float* row = x0cat + (size_t)b * K0;
  for (int d = tid; d < HH; d += 256) {
    float        acc = 0.f;
    const float* eb  = enc + (size_t)b * SS * HH + d;
#pragma unroll 8
    for (int s = 0; s < SS; ++s) acc += sc[s] * eb[(size_t)s * HH];
    row[d] = acc;
  }
  for (int e = tid; e < EE; e += 256)
    row[HH + e] = x_in[((size_t)b * TT + t) * EE + e];
  for (int d = tid; d < HH; d += 256)
    row[HH + EE + d] = h0[(size_t)b * HH + d];
}

// ---------------------------------------------------------------------------
__global__ void lstm_gates0(const float* __restrict__ z, float* __restrict__ h0,
                            float* __restrict__ c0, const float* __restrict__ h1old,
                            float* __restrict__ x1cat) {
  const int i = blockIdx.x * blockDim.x + threadIdx.x;
  if (i >= BB * HH) return;
  const int    b  = i >> 9, j = i & (HH - 1);
  const float* zb = z + (size_t)b * 4 * HH;
  const float  iv = sigmoidf_(zb[j]);
  const float  fv = sigmoidf_(zb[HH + j]);
  const float  gv = tanhf(zb[2 * HH + j]);
  const float  ov = sigmoidf_(zb[3 * HH + j]);
  const float  c  = fv * c0[i] + iv * gv;
  const float  hv = ov * tanhf(c);
  c0[i] = c;
  h0[i] = hv;
  x1cat[(size_t)b * (2 * HH) + j]      = hv;        // new h0
  x1cat[(size_t)b * (2 * HH) + HH + j] = h1old[i];  // old h1 (recurrent)
}

__global__ void lstm_gates1(const float* __restrict__ z, float* __restrict__ h1,
                            float* __restrict__ c1, __bf16* __restrict__ Abuf) {
  const int i = blockIdx.x * blockDim.x + threadIdx.x;
  if (i >= BB * HH) return;
  const int    b  = i >> 9, j = i & (HH - 1);
  const float* zb = z + (size_t)b * 4 * HH;
  const float  iv = sigmoidf_(zb[j]);
  const float  fv = sigmoidf_(zb[HH + j]);
  const float  gv = tanhf(zb[2 * HH + j]);
  const float  ov = sigmoidf_(zb[3 * HH + j]);
  const float  c  = fv * c1[i] + iv * gv;
  const float  hv = ov * tanhf(c);
  c1[i] = c;
  h1[i] = hv;
  // pack into bf16 WMMA A-fragment order: (m=b, k=j) -> slot
  const int kk   = j >> 5;
  const int hh   = (j >> 3) & 1;
  const int jj   = (j & 31) - 8 * hh;
  const int j16  = (jj < 8) ? jj : jj - 8;
  const int lane = (b & 15) + 16 * hh;
  const int tm   = b >> 4;
  Abuf[(((kk * 2 + tm) * 32 + lane) << 4) + j16] = (__bf16)hv;
}

// ---------------------------------------------------------------------------
__global__ void init_state(const float* __restrict__ hidden0, const float* __restrict__ cell0,
                           float* __restrict__ h0, float* __restrict__ c0,
                           float* __restrict__ h1, float* __restrict__ c1) {
  const int i = blockIdx.x * blockDim.x + threadIdx.x;
  if (i >= BB * HH) return;
  h0[i] = hidden0[i];
  h1[i] = hidden0[BB * HH + i];
  c0[i] = cell0[i];
  c1[i] = cell0[BB * HH + i];
}

__global__ void concat_rows(const float* __restrict__ A, int rowsA,
                            const float* __restrict__ Bm, int rowsB, int ncols,
                            float* __restrict__ dst) {
  const size_t i     = (size_t)blockIdx.x * blockDim.x + threadIdx.x;
  const size_t total = (size_t)(rowsA + rowsB) * ncols;
  if (i >= total) return;
  const int r = (int)(i / ncols), c = (int)(i % ncols);
  dst[i] = (r < rowsA) ? A[(size_t)r * ncols + c] : Bm[(size_t)(r - rowsA) * ncols + c];
}

__global__ void f32_to_bf16_vec4(const float* __restrict__ src, __bf16* __restrict__ dst,
                                 int n4) {
  const int i = blockIdx.x * blockDim.x + threadIdx.x;
  if (i >= n4) return;
  const float4 v = ((const float4*)src)[i];
  dst[4 * i + 0] = (__bf16)v.x;
  dst[4 * i + 1] = (__bf16)v.y;
  dst[4 * i + 2] = (__bf16)v.z;
  dst[4 * i + 3] = (__bf16)v.w;
}

__global__ void finalize_state(const float* __restrict__ h0, const float* __restrict__ h1,
                               const float* __restrict__ c0, const float* __restrict__ c1,
                               float* __restrict__ out) {
  const int i = blockIdx.x * blockDim.x + threadIdx.x;
  if (i >= BB * HH) return;
  float* p           = out + (size_t)BB * TT * DVV;
  p[i]               = h0[i];
  p[BB * HH + i]     = h1[i];
  p[2 * BB * HH + i] = c0[i];
  p[3 * BB * HH + i] = c1[i];
}

// ---------------------------------------------------------------------------
extern "C" void kernel_launch(void* const* d_in, const int* in_sizes, int n_in,
                              void* d_out, int out_size, void* d_ws, size_t ws_size,
                              hipStream_t stream) {
  (void)in_sizes; (void)n_in; (void)out_size; (void)ws_size;
  const float* initial_input = (const float*)d_in[0];
  const float* hidden0       = (const float*)d_in[1];
  const float* cell0         = (const float*)d_in[2];
  const float* enc           = (const float*)d_in[3];
  const float* W1w           = (const float*)d_in[4];
  const float* b1            = (const float*)d_in[5];
  const float* W2w           = (const float*)d_in[6];
  const float* b2            = (const float*)d_in[7];
  const float* Vv            = (const float*)d_in[8];
  const float* bV            = (const float*)d_in[9];
  const float* Wx0           = (const float*)d_in[10];
  const float* Wh0           = (const float*)d_in[11];
  const float* bL0           = (const float*)d_in[12];
  const float* Wx1           = (const float*)d_in[13];
  const float* Wh1           = (const float*)d_in[14];
  const float* bL1           = (const float*)d_in[15];
  const float* Wd            = (const float*)d_in[16];
  const float* bd            = (const float*)d_in[17];
  float*       out           = (float*)d_out;

  float*  ws       = (float*)d_ws;
  float*  enc_proj = ws;                  // 1,048,576 f32
  float*  q        = enc_proj + 1048576;  // 16,384
  float*  h0       = q + 16384;
  float*  c0       = h0 + 16384;
  float*  h1       = c0 + 16384;
  float*  c1       = h1 + 16384;
  float*  x0cat    = c1 + 16384;       // 32*1280
  float*  x1cat    = x0cat + BB * K0;  // 32*1024
  float*  z0       = x1cat + 32768;    // 32*2048
  float*  z1       = z0 + 65536;       // 32*2048
  float*  W0cat    = z1 + 65536;       // 1280*2048
  float*  W1cat    = W0cat + (size_t)K0 * 4 * HH;  // 1024*2048
  __bf16* Wdbf     = (__bf16*)(W1cat + (size_t)2 * HH * 4 * HH);
  __bf16* Abuf     = Wdbf + (size_t)HH * DVV;

  // per-launch setup
  init_state<<<64, 256, 0, stream>>>(hidden0, cell0, h0, c0, h1, c1);
  concat_rows<<<((K0)*4 * HH + 255) / 256, 256, 0, stream>>>(Wx0, HH + EE, Wh0, HH,
                                                             4 * HH, W0cat);
  concat_rows<<<(2 * HH * 4 * HH + 255) / 256, 256, 0, stream>>>(Wx1, HH, Wh1, HH,
                                                                 4 * HH, W1cat);
  f32_to_bf16_vec4<<<(HH * DVV / 4 + 255) / 256, 256, 0, stream>>>(Wd, Wdbf,
                                                                   HH * DVV / 4);
  gemm_f32_wmma<<<(BB * SS / 16) * (AAT / 16) / 8, 256, 0, stream>>>(
      enc, W2w, b2, enc_proj, BB * SS, AAT, HH);

  // sequential decode
  for (int t = 0; t < TT; ++t) {
    gemm_f32_wmma_m32<<<4, 256, 0, stream>>>(h1, W1w, b1, q, AAT, HH);
    attn_ctx<<<BB, 256, 0, stream>>>(q, enc_proj, enc, Vv, bV, initial_input, h0, t,
                                     x0cat);
    gemm_f32_wmma_m32<<<16, 256, 0, stream>>>(x0cat, W0cat, bL0, z0, 4 * HH, K0);
    lstm_gates0<<<64, 256, 0, stream>>>(z0, h0, c0, h1, x1cat);
    gemm_f32_wmma_m32<<<16, 256, 0, stream>>>(x1cat, W1cat, bL1, z1, 4 * HH, 2 * HH);
    lstm_gates1<<<64, 256, 0, stream>>>(z1, h1, c1, Abuf);
    gemm_logits_bf16<<<250, 256, 0, stream>>>(Abuf, Wdbf, bd, out, t);
  }
  finalize_state<<<64, 256, 0, stream>>>(h0, h1, c0, c1, out);
}
